// SAABAttention_9792525435170
// MI455X (gfx1250) — compile-verified, matrix operand order checked by reading, not run on hardware
//
#include <hip/hip_runtime.h>

typedef __attribute__((ext_vector_type(16))) __bf16 v16bf;
typedef __attribute__((ext_vector_type(8)))  float  v8f;

typedef unsigned short u16;
typedef unsigned int   u32;

#define B_   2
#define S_   2048
#define D_   1024
#define H_   16
#define HD_  64
#define SCALE_ 0.125f     /* 64^-0.5 */

union Frag {
  v16bf v;
  uint4 u[2];
};

__device__ __forceinline__ u16 f2bf(float x) {
  u32 u = __float_as_uint(x);
  u += 0x7fffu + ((u >> 16) & 1u);       // round-to-nearest-even
  return (u16)(u >> 16);
}
__device__ __forceinline__ u32 pk2(float a, float b) {
  return (u32)f2bf(a) | ((u32)f2bf(b) << 16);
}

__device__ __forceinline__ v8f wmma_bf16(const Frag& a, const Frag& b, v8f c) {
  return __builtin_amdgcn_wmma_f32_16x16x32_bf16(
      false, a.v, false, b.v, (short)0, c, false, false);
}

// unpack 8 bf16 (packed 2/dword) to f32: even -> low half, odd -> high half
__device__ __forceinline__ void unpack8(uint4 w, float* f) {
  f[0] = __uint_as_float(w.x << 16); f[1] = __uint_as_float(w.x & 0xffff0000u);
  f[2] = __uint_as_float(w.y << 16); f[3] = __uint_as_float(w.y & 0xffff0000u);
  f[4] = __uint_as_float(w.z << 16); f[5] = __uint_as_float(w.z & 0xffff0000u);
  f[6] = __uint_as_float(w.w << 16); f[7] = __uint_as_float(w.w & 0xffff0000u);
}

// structural bias from packed ids:
// w0 = field | entity<<8 | time<<16 | tokentype<<24
// w1 = edge  | role<<8   | (field==PAD)<<16
__device__ __forceinline__ float sbias(u32 q0, u32 q1, u32 k0, u32 k1) {
  u32 x0 = q0 ^ k0;
  u32 x1 = q1 ^ k1;
  float b = 0.f;
  b += (x0 & 0x000000ffu) ? 0.f : 1.0f;                 // field eq
  b += (x0 & 0x0000ff00u) ? 0.f : 1.0f;                 // entity eq
  b += (x0 & 0x00ff0000u) ? 0.f : 0.5f;                 // time eq
  b += (x0 & 0xff000000u) ? 0.f : 0.3f;                 // token_type eq
  b += (((q1 | k1) & 0x000000ffu) != 0u) ? 1.5f : 0.f;  // has_edge
  b += (x1 & 0x0000ff00u) ? 0.f : 0.5f;                 // role eq
  if (((q1 | k1) & 0x10000u) != 0u) b = -1.0e30f;       // PAD mask
  return b;
}

__global__ __launch_bounds__(256)
void pack_ids(const int* __restrict__ f, const int* __restrict__ e,
              const int* __restrict__ t, const int* __restrict__ tt,
              const int* __restrict__ ed, const int* __restrict__ ro,
              u32* __restrict__ p0, u32* __restrict__ p1, int n) {
  int i = blockIdx.x * blockDim.x + threadIdx.x;
  if (i >= n) return;
  u32 w0 = (u32)(f[i] & 0xff) | ((u32)(e[i] & 0xff) << 8) |
           ((u32)(t[i] & 0xff) << 16) | ((u32)(tt[i] & 0xff) << 24);
  u32 w1 = (u32)(ed[i] & 0xff) | ((u32)(ro[i] & 0xff) << 8) |
           ((f[i] == 0) ? 0x10000u : 0u);
  p0[i] = w0;
  p1[i] = w1;
}

// Bias table in WMMA-C-layout-friendly tiling: bias_t[B][S/8][S][8] bf16.
// Each thread: one key column x one 8-query group -> one 16B store.
__global__ __launch_bounds__(256)
void build_bias(const u32* __restrict__ p0, const u32* __restrict__ p1,
                u16* __restrict__ bias) {
  int idx = blockIdx.x * 256 + threadIdx.x;       // B*S*S/8 threads
  int kk  = idx & (S_ - 1);
  int qg  = (idx >> 11) & (S_ / 8 - 1);
  int b   = idx >> 19;
  const u32* pb0 = p0 + (size_t)b * S_;
  const u32* pb1 = p1 + (size_t)b * S_;
  u32 k0 = pb0[kk], k1 = pb1[kk];
  uint4 qa0 = *(const uint4*)(pb0 + qg * 8);
  uint4 qa1 = *(const uint4*)(pb0 + qg * 8 + 4);
  uint4 qb0 = *(const uint4*)(pb1 + qg * 8);
  uint4 qb1 = *(const uint4*)(pb1 + qg * 8 + 4);
  uint4 o;
  o.x = pk2(sbias(qa0.x, qb0.x, k0, k1), sbias(qa0.y, qb0.y, k0, k1));
  o.y = pk2(sbias(qa0.z, qb0.z, k0, k1), sbias(qa0.w, qb0.w, k0, k1));
  o.z = pk2(sbias(qa1.x, qb1.x, k0, k1), sbias(qa1.y, qb1.y, k0, k1));
  o.w = pk2(sbias(qa1.z, qb1.z, k0, k1), sbias(qa1.w, qb1.w, k0, k1));
  *(uint4*)&bias[((size_t)(b * (S_ / 8) + qg) * S_ + kk) * 8] = o;
}

// fp32 -> bf16 bulk convert (4 elements / thread)
__global__ __launch_bounds__(256)
void cvt_bf16(const float* __restrict__ src, u16* __restrict__ dst, int n4) {
  int i = blockIdx.x * blockDim.x + threadIdx.x;
  if (i >= n4) return;
  float4 v = ((const float4*)src)[i];
  ((uint2*)dst)[i] = make_uint2(pk2(v.x, v.y), pk2(v.z, v.w));
}

// ---------------------------------------------------------------------------
// GEMM: out[m][n] = A[m][:] . W[n][:] + bias[n]   (A @ W^T + b), all bf16 in.
// No LDS: each wave loads its A/B fragments straight from global (L2-served).
// MODE 0: outputs bf16 Q[B,H,S,64]/K[B,H,S,64]/Vt[B,H,64,S], z picks matrix.
// MODE 1: result fp32 into outF[B*S, D].
// block = 128 threads (4 waves), tile 64(M) x 64(N), k-step 32.
// ---------------------------------------------------------------------------
template<int MODE>
__global__ __launch_bounds__(128)
void gemm_bf16(const u16* __restrict__ A0, const u16* __restrict__ A1,
               const u16* __restrict__ A2,
               const u16* __restrict__ W0, const u16* __restrict__ W1,
               const u16* __restrict__ W2,
               const float* __restrict__ b0, const float* __restrict__ b1,
               const float* __restrict__ b2,
               u16* __restrict__ outQ, u16* __restrict__ outK,
               u16* __restrict__ outVt, float* __restrict__ outF) {
  const int lane  = threadIdx.x & 31;
  const int wave  = threadIdx.x >> 5;
  const int n16   = lane & 15;
  const int hi    = lane >> 4;
  const int kb    = hi * 8;
  const int mbase = blockIdx.x * 64;
  const int nbase = blockIdx.y * 64;
  const int z     = blockIdx.z;

  const u16* A    = (z == 0) ? A0 : (z == 1) ? A1 : A2;
  const u16* W    = (z == 0) ? W0 : (z == 1) ? W1 : W2;
  const float* bias = (z == 0) ? b0 : (z == 1) ? b1 : b2;

  const u16* ar  = A + (size_t)(mbase + wave * 16 + n16) * D_;
  const u16* wr0 = W + (size_t)(nbase + n16) * D_;
  const u16* wr1 = W + (size_t)(nbase + 16 + n16) * D_;
  const u16* wr2 = W + (size_t)(nbase + 32 + n16) * D_;
  const u16* wr3 = W + (size_t)(nbase + 48 + n16) * D_;

  v8f acc[4];
#pragma unroll
  for (int t = 0; t < 4; ++t)
#pragma unroll
    for (int i = 0; i < 8; ++i) acc[t][i] = 0.f;

  for (int kk = 0; kk < D_; kk += 32) {
    Frag fa;
    fa.u[0] = *(const uint4*)(ar + kk + kb);
    fa.u[1] = *(const uint4*)(ar + kk + 16 + kb);
    Frag fb0, fb1, fb2, fb3;
    fb0.u[0] = *(const uint4*)(wr0 + kk + kb);
    fb0.u[1] = *(const uint4*)(wr0 + kk + 16 + kb);
    fb1.u[0] = *(const uint4*)(wr1 + kk + kb);
    fb1.u[1] = *(const uint4*)(wr1 + kk + 16 + kb);
    fb2.u[0] = *(const uint4*)(wr2 + kk + kb);
    fb2.u[1] = *(const uint4*)(wr2 + kk + 16 + kb);
    fb3.u[0] = *(const uint4*)(wr3 + kk + kb);
    fb3.u[1] = *(const uint4*)(wr3 + kk + 16 + kb);
    acc[0] = wmma_bf16(fa, fb0, acc[0]);
    acc[1] = wmma_bf16(fa, fb1, acc[1]);
    acc[2] = wmma_bf16(fa, fb2, acc[2]);
    acc[3] = wmma_bf16(fa, fb3, acc[3]);
  }

#pragma unroll
  for (int t = 0; t < 4; ++t) {
    const int cn   = nbase + t * 16 + n16;
    const float bv = bias[cn];
#pragma unroll
    for (int r = 0; r < 8; ++r) {
      const int gm = mbase + wave * 16 + r + 8 * hi;
      float val = acc[t][r] + bv;
      if (MODE == 0) {
        int bb = gm >> 11, ss = gm & (S_ - 1);
        int hh = cn >> 6,  hd = cn & (HD_ - 1);
        u16 o = f2bf(val);
        if (z == 0)      outQ [(((size_t)bb * H_ + hh) * S_ + ss) * HD_ + hd] = o;
        else if (z == 1) outK [(((size_t)bb * H_ + hh) * S_ + ss) * HD_ + hd] = o;
        else             outVt[(((size_t)bb * H_ + hh) * HD_ + hd) * S_ + ss] = o;
      } else {
        outF[(size_t)gm * D_ + cn] = val;
      }
    }
  }
}

// ---------------------------------------------------------------------------
// Flash attention with precomputed structural bias table (C-layout tiling).
// block = 128 threads (4 waves); each wave owns 16 queries, block owns 64.
// Per 32-key block: 4 score WMMAs + 4 PV WMMAs + 1 row-sum WMMA (P @ ones),
// 2 coalesced b128 bias loads.
// ---------------------------------------------------------------------------
__global__ __launch_bounds__(128)
void attn_flash(const u16* __restrict__ Q, const u16* __restrict__ K,
                const u16* __restrict__ Vt, const u16* __restrict__ Bias,
                u16* __restrict__ Ob) {
  __shared__ u16 Plds[4][16 * 40];   // per-wave P relayout buffer (80B pitch)

  const int lane  = threadIdx.x & 31;
  const int wave  = threadIdx.x >> 5;
  const int n16   = lane & 15;
  const int hi    = lane >> 4;
  const int kb    = hi * 8;
  const int h     = blockIdx.y;
  const int b     = blockIdx.z;
  const int qbase = blockIdx.x * 64 + wave * 16;

  const u16* Qh = Q  + (size_t)(b * H_ + h) * S_ * HD_;
  const u16* Kh = K  + (size_t)(b * H_ + h) * S_ * HD_;
  const u16* Vh = Vt + (size_t)(b * H_ + h) * HD_ * S_;
  // bias row-group base: this lane's 8 C-layout rows = one contiguous group
  const u16* Br = Bias + (size_t)(b * (S_ / 8) + ((qbase + 8 * hi) >> 3)) * S_ * 8;

  // Q fragments (A layout: lane row m = lane&15, K split by half-wave)
  Frag fq0, fq1;
  {
    const u16* qr = Qh + (size_t)(qbase + n16) * HD_;
    fq0.u[0] = *(const uint4*)(qr + kb);
    fq0.u[1] = *(const uint4*)(qr + 16 + kb);
    fq1.u[0] = *(const uint4*)(qr + 32 + kb);
    fq1.u[1] = *(const uint4*)(qr + 48 + kb);
  }

  // all-ones B fragment for row-sum accumulation (bf16 1.0 = 0x3f80)
  Frag fone;
  fone.u[0] = make_uint4(0x3f803f80u, 0x3f803f80u, 0x3f803f80u, 0x3f803f80u);
  fone.u[1] = fone.u[0];

  float rm[8];
  v8f acc[4], accl;
#pragma unroll
  for (int r = 0; r < 8; ++r) rm[r] = -3.0e38f;
#pragma unroll
  for (int i = 0; i < 8; ++i) accl[i] = 0.f;
#pragma unroll
  for (int t = 0; t < 4; ++t)
#pragma unroll
    for (int i = 0; i < 8; ++i) acc[t][i] = 0.f;

  u16* Pw = &Plds[wave][0];

  for (int k0 = 0; k0 < S_; k0 += 32) {
    // prefetch next key block (K rows + Vt columns)
    if (k0 + 32 < S_) {
      __builtin_prefetch(Kh + (size_t)(k0 + 32 + lane) * HD_, 0, 1);
      __builtin_prefetch(Vh + (size_t)lane * S_ + k0 + 32, 0, 1);
      __builtin_prefetch(Vh + (size_t)(lane + 32) * S_ + k0 + 32, 0, 1);
    }

    // bias: 2 coalesced b128 loads cover all 16 score elements of this lane
    uint4 bw0 = *(const uint4*)(Br + (size_t)(k0 + n16) * 8);
    uint4 bw1 = *(const uint4*)(Br + (size_t)(k0 + 16 + n16) * 8);

    // ---- scores: S = Q . K^T (B-frag lane col n = key row n) ----
    Frag fk00, fk01, fk10, fk11;
    {
      const u16* kr0 = Kh + (size_t)(k0 + n16) * HD_;
      const u16* kr1 = Kh + (size_t)(k0 + 16 + n16) * HD_;
      fk00.u[0] = *(const uint4*)(kr0 + kb);      fk00.u[1] = *(const uint4*)(kr0 + 16 + kb);
      fk01.u[0] = *(const uint4*)(kr0 + 32 + kb); fk01.u[1] = *(const uint4*)(kr0 + 48 + kb);
      fk10.u[0] = *(const uint4*)(kr1 + kb);      fk10.u[1] = *(const uint4*)(kr1 + 16 + kb);
      fk11.u[0] = *(const uint4*)(kr1 + 32 + kb); fk11.u[1] = *(const uint4*)(kr1 + 48 + kb);
    }
    v8f s0, s1;
#pragma unroll
    for (int i = 0; i < 8; ++i) { s0[i] = 0.f; s1[i] = 0.f; }
    s0 = wmma_bf16(fq0, fk00, s0);
    s0 = wmma_bf16(fq1, fk01, s0);
    s1 = wmma_bf16(fq0, fk10, s1);
    s1 = wmma_bf16(fq1, fk11, s1);

    // ---- add bias, online softmax ----
    float bv0[8], bv1[8];
    unpack8(bw0, bv0);
    unpack8(bw1, bv1);
    float v0[8], v1[8], bmax[8], al[8];
#pragma unroll
    for (int r = 0; r < 8; ++r) {
      v0[r] = s0[r] * SCALE_ + bv0[r];
      v1[r] = s1[r] * SCALE_ + bv1[r];
      bmax[r] = fmaxf(v0[r], v1[r]);
    }
    // row max across the 16 lanes of each half (mask < 16 stays in half)
#pragma unroll
    for (int mask = 1; mask < 16; mask <<= 1)
#pragma unroll
      for (int r = 0; r < 8; ++r)
        bmax[r] = fmaxf(bmax[r], __shfl_xor(bmax[r], mask, 32));
#pragma unroll
    for (int r = 0; r < 8; ++r) {
      float mn = fmaxf(rm[r], bmax[r]);
      al[r] = __expf(rm[r] - mn);
      rm[r] = mn;
      v0[r] = __expf(v0[r] - mn);
      v1[r] = __expf(v1[r] - mn);
    }
#pragma unroll
    for (int t = 0; t < 4; ++t)
#pragma unroll
      for (int r = 0; r < 8; ++r) acc[t][r] *= al[r];
#pragma unroll
    for (int r = 0; r < 8; ++r) accl[r] *= al[r];

    // ---- relayout P: C layout -> A layout through per-wave LDS tile ----
#pragma unroll
    for (int r = 0; r < 8; ++r) {
      int row = r + 8 * hi;
      Pw[row * 40 + n16]      = f2bf(v0[r]);
      Pw[row * 40 + 16 + n16] = f2bf(v1[r]);
    }
    asm volatile("s_wait_dscnt 0" ::: "memory");

    Frag fp;
    const u16* pr = Pw + n16 * 40;
    fp.u[0] = *(const uint4*)(pr + kb);
    fp.u[1] = *(const uint4*)(pr + 16 + kb);

    // ---- acc += P @ V ; accl += P @ ones (row sums, same alpha scaling) ----
#pragma unroll
    for (int t = 0; t < 4; ++t) {
      Frag fv;
      const u16* vr = Vh + (size_t)(t * 16 + n16) * S_ + k0;
      fv.u[0] = *(const uint4*)(vr + kb);
      fv.u[1] = *(const uint4*)(vr + 16 + kb);
      acc[t] = wmma_bf16(fp, fv, acc[t]);
    }
    accl = wmma_bf16(fp, fone, accl);
  }

  // normalize (rcp once per row) and store bf16 attention output [B,S,D]
  float inv[8];
#pragma unroll
  for (int r = 0; r < 8; ++r) inv[r] = __builtin_amdgcn_rcpf(accl[r]);
#pragma unroll
  for (int t = 0; t < 4; ++t)
#pragma unroll
    for (int r = 0; r < 8; ++r) {
      float val = acc[t][r] * inv[r];
      int srow = qbase + r + 8 * hi;
      Ob[((size_t)b * S_ + srow) * D_ + h * HD_ + t * 16 + n16] = f2bf(val);
    }
}

// ---------------------------------------------------------------------------
extern "C" void kernel_launch(void* const* d_in, const int* in_sizes, int n_in,
                              void* d_out, int out_size, void* d_ws, size_t ws_size,
                              hipStream_t stream) {
  const float* q   = (const float*)d_in[0];
  const float* k   = (const float*)d_in[1];
  const float* v   = (const float*)d_in[2];
  const int* fid   = (const int*)d_in[3];
  const int* eid   = (const int*)d_in[4];
  const int* tmid  = (const int*)d_in[5];
  const int* ttid  = (const int*)d_in[6];
  const int* edid  = (const int*)d_in[7];
  const int* roid  = (const int*)d_in[8];
  const float* Wq  = (const float*)d_in[9];
  const float* bq  = (const float*)d_in[10];
  const float* Wk  = (const float*)d_in[11];
  const float* bk  = (const float*)d_in[12];
  const float* Wv  = (const float*)d_in[13];
  const float* bv  = (const float*)d_in[14];
  const float* Wo  = (const float*)d_in[15];
  const float* bo  = (const float*)d_in[16];
  float* out = (float*)d_out;

  const size_t nTok = (size_t)B_ * S_;   // 4096
  const size_t nAct = nTok * D_;         // 4,194,304
  const size_t nW   = (size_t)D_ * D_;   // 1,048,576
  const size_t nBias = (size_t)B_ * S_ * S_;  // 8,388,608

  char* ws = (char*)d_ws;
  u16* Qb   = (u16*)ws;
  u16* Kb   = Qb + nAct;
  u16* Vtb  = Kb + nAct;
  u16* Ob   = Vtb + nAct;
  u16* Bias = Ob + nAct;
  u16* qbf  = Bias + nBias;
  u16* kbf  = qbf + nAct;
  u16* vbf  = kbf + nAct;
  u16* Wqb  = vbf + nAct;
  u16* Wkb  = Wqb + nW;
  u16* Wvb  = Wkb + nW;
  u16* Wob  = Wvb + nW;
  u32* p0   = (u32*)(Wob + nW);
  u32* p1   = p0 + nTok;

  pack_ids<<<dim3((unsigned)((nTok + 255) / 256)), dim3(256), 0, stream>>>(
      fid, eid, tmid, ttid, edid, roid, p0, p1, (int)nTok);

  build_bias<<<dim3((unsigned)(nBias / 8 / 256)), dim3(256), 0, stream>>>(p0, p1, Bias);

  cvt_bf16<<<dim3((unsigned)(nAct / 4 / 256)), dim3(256), 0, stream>>>(q, qbf, (int)(nAct / 4));
  cvt_bf16<<<dim3((unsigned)(nAct / 4 / 256)), dim3(256), 0, stream>>>(k, kbf, (int)(nAct / 4));
  cvt_bf16<<<dim3((unsigned)(nAct / 4 / 256)), dim3(256), 0, stream>>>(v, vbf, (int)(nAct / 4));
  cvt_bf16<<<dim3((unsigned)(nW / 4 / 256)), dim3(256), 0, stream>>>(Wq, Wqb, (int)(nW / 4));
  cvt_bf16<<<dim3((unsigned)(nW / 4 / 256)), dim3(256), 0, stream>>>(Wk, Wkb, (int)(nW / 4));
  cvt_bf16<<<dim3((unsigned)(nW / 4 / 256)), dim3(256), 0, stream>>>(Wv, Wvb, (int)(nW / 4));
  cvt_bf16<<<dim3((unsigned)(nW / 4 / 256)), dim3(256), 0, stream>>>(Wo, Wob, (int)(nW / 4));

  // Q / K / Vt projections (grid.z = 0,1,2)
  gemm_bf16<0><<<dim3(64, 16, 3), dim3(128), 0, stream>>>(
      qbf, kbf, vbf, Wqb, Wkb, Wvb, bq, bk, bv, Qb, Kb, Vtb, nullptr);

  attn_flash<<<dim3(32, 16, 2), dim3(128), 0, stream>>>(Qb, Kb, Vtb, Bias, Ob);

  // output projection -> fp32 d_out
  gemm_bf16<1><<<dim3(64, 16, 1), dim3(128), 0, stream>>>(
      Ob, nullptr, nullptr, Wob, nullptr, nullptr, bo, nullptr, nullptr,
      nullptr, nullptr, nullptr, out);
}
